// MLAAttention_32530082299945
// MI455X (gfx1250) — compile-verified, matrix operand order checked by reading, not run on hardware
//
#include <hip/hip_runtime.h>
#include <hip/hip_bf16.h>

#define T_SEQ   2048
#define HIDDEN  4096
#define NH      32
#define QK_NOPE 128
#define QK_ROPE 64
#define DQK     192   // QK_NOPE + QK_ROPE
#define DV      128
#define Q_LORA  1536
#define KV_LORA 512
#define QKVA_N  2112  // Q_LORA + KV_LORA + QK_ROPE
#define ATT_SCALE 0.07216878364870323f  // 1/sqrt(192)

typedef __attribute__((ext_vector_type(16))) __bf16       v16bf;
typedef __attribute__((ext_vector_type(8)))  float        f32x8;
typedef __attribute__((ext_vector_type(4)))  unsigned int u32x4;

__device__ __forceinline__ unsigned short f2bf(float f) {
    unsigned int u = __float_as_uint(f);
    unsigned int r = 0x7FFFu + ((u >> 16) & 1u);
    return (unsigned short)((u + r) >> 16);
}

// Load a 16-bit WMMA operand fragment: two contiguous 8-element (16B) runs
// at p[0..7] and p[16..23], matching the CDNA5 16-bit A/B lane packing
// (caller supplies p = row_base + khalf*8, khalf = lane>=16).
__device__ __forceinline__ v16bf load_frag(const unsigned short* p) {
    union { v16bf v; u32x4 u[2]; } f;
    f.u[0] = *(const u32x4*)(p);
    f.u[1] = *(const u32x4*)(p + 16);
    return f.v;
}

__device__ __forceinline__ f32x8 fzero8() {
    f32x8 z = {0.f, 0.f, 0.f, 0.f, 0.f, 0.f, 0.f, 0.f};
    return z;
}

// CDNA5 async DMA: 16B per lane, global -> LDS, tracked by ASYNCcnt.
// Generic LDS pointer's low 32 bits are the LDS byte address (aperture rule).
__device__ __forceinline__ void async_copy_b128(const void* gptr, void* lptr) {
    unsigned           lds = (unsigned)(size_t)lptr;
    unsigned long long ga  = (unsigned long long)(size_t)gptr;
    asm volatile("global_load_async_to_lds_b128 %0, %1, off"
                 :: "v"(lds), "v"(ga) : "memory");
}
__device__ __forceinline__ void wait_async0() {
    asm volatile("s_wait_asynccnt 0" ::: "memory");
}

// ---------------------------------------------------------------------------
// fp32 -> bf16 bulk conversion (row-major, activations)
// ---------------------------------------------------------------------------
__global__ void k_f32_to_bf16(const float* __restrict__ in,
                              unsigned short* __restrict__ out, long n) {
    long i = (long)blockIdx.x * blockDim.x + threadIdx.x;
    long stride = (long)gridDim.x * blockDim.x;
    for (; i < n; i += stride) out[i] = f2bf(in[i]);
}

// ---------------------------------------------------------------------------
// fp32 [R][C] -> bf16 transposed [C][R], tiled 32x32 through LDS (weights).
// Requires R%32==0, C%32==0. Block (32,8).
// ---------------------------------------------------------------------------
__global__ __launch_bounds__(256)
void k_f32_to_bf16_T(const float* __restrict__ in,
                     unsigned short* __restrict__ out, int R, int C) {
    __shared__ float tile[32][33];
    const int cb = blockIdx.x * 32, rb = blockIdx.y * 32;
#pragma unroll
    for (int i = threadIdx.y; i < 32; i += 8)
        tile[i][threadIdx.x] = in[(size_t)(rb + i) * C + cb + threadIdx.x];
    __syncthreads();
#pragma unroll
    for (int i = threadIdx.y; i < 32; i += 8)
        out[(size_t)(cb + i) * R + rb + threadIdx.x] = f2bf(tile[threadIdx.x][i]);
}

// ---------------------------------------------------------------------------
// bf16 GEMM, TN form: C[M,N] (f32) = A[M,K] @ Bt[N,K]^T (both row-major bf16).
// Block tile 256x64 (8 waves, each 64x32). K staged 64-deep, double-buffered
// via global_load_async_to_lds_b128 (ASYNCcnt) -> 16 WMMAs per barrier.
// Requires M%256==0, N%64==0, K%64==0.
// ---------------------------------------------------------------------------
__global__ __launch_bounds__(256)
void k_gemm_bf16_tn(const unsigned short* __restrict__ A,
                    const unsigned short* __restrict__ Bt,
                    float* __restrict__ C, int M, int N, int K) {
    __shared__ __align__(16) unsigned short As[2][256 * 80];  // [m][k64] pad->80
    __shared__ __align__(16) unsigned short Bs[2][64 * 80];   // [n][k64] pad->80

    const int tid  = threadIdx.x;
    const int wave = tid >> 5, lane = tid & 31;
    const int lm = lane & 15, lh = lane >> 4;
    const int wm = wave >> 1, wn = wave & 1;          // 4 x 2 wave grid
    const int mbase = blockIdx.x * 256;
    const int nbase = blockIdx.y * 64;

    auto stage = [&](int buf, int kk) {
#pragma unroll
        for (int i = 0; i < 8; ++i) {                 // A: 2048 x 16B chunks
            int cc = tid + i * 256, row = cc >> 2, kc = (cc & 3) * 8;
            async_copy_b128(&A[(size_t)(mbase + row) * K + kk + kc],
                            &As[buf][row * 80 + kc]);
        }
        {                                             // B: 256 x 16B chunks
            int row = tid >> 2, kc = (tid & 3) * 8;
            async_copy_b128(&Bt[(size_t)(nbase + row) * K + kk + kc],
                            &Bs[buf][row * 80 + kc]);
        }
    };

    f32x8 acc[4][2];
#pragma unroll
    for (int r = 0; r < 4; ++r)
#pragma unroll
        for (int c = 0; c < 2; ++c) acc[r][c] = fzero8();

    stage(0, 0);
    wait_async0();
    __syncthreads();

    int buf = 0;
    for (int kk = 0; kk < K; kk += 64) {
        if (kk + 64 < K) stage(buf ^ 1, kk + 64);     // prefetch next stage
#pragma unroll
        for (int ks = 0; ks < 2; ++ks) {              // two K=32 WMMA steps
            v16bf af[4], bf_[2];
#pragma unroll
            for (int r = 0; r < 4; ++r)
                af[r] = load_frag(
                    &As[buf][(wm * 64 + r * 16 + lm) * 80 + ks * 32 + lh * 8]);
#pragma unroll
            for (int c = 0; c < 2; ++c)
                bf_[c] = load_frag(
                    &Bs[buf][(wn * 32 + c * 16 + lm) * 80 + ks * 32 + lh * 8]);
#pragma unroll
            for (int r = 0; r < 4; ++r)
#pragma unroll
                for (int c = 0; c < 2; ++c)
                    acc[r][c] = __builtin_amdgcn_wmma_f32_16x16x32_bf16(
                        false, af[r], false, bf_[c], (short)0, acc[r][c],
                        false, false);
        }
        wait_async0();          // our async writes into buf^1 have landed
        __syncthreads();        // publish to all waves / all done reading buf
        buf ^= 1;
    }

    // C layout: vgpr e -> row = e + 8*lh, col = lane&15. Incremental offsets.
    float* Cp = C + (size_t)(mbase + wm * 64 + 8 * lh) * N
                  + nbase + wn * 32 + lm;
#pragma unroll
    for (int r = 0; r < 4; ++r)
#pragma unroll
        for (int c = 0; c < 2; ++c)
#pragma unroll
            for (int e = 0; e < 8; ++e)
                Cp[(size_t)(r * 16 + e) * N + c * 16] = acc[r][c][e];
}

// ---------------------------------------------------------------------------
// RMSNorm row kernel: out_bf16[t, 0:len] = x * rsqrt(mean(x^2)+eps) * w
// ---------------------------------------------------------------------------
__global__ __launch_bounds__(256)
void k_rmsnorm_bf16(const float* __restrict__ in, int stride, int off,
                    const float* __restrict__ w,
                    unsigned short* __restrict__ out, int len) {
    __shared__ float red[256];
    const int t = blockIdx.x;
    const float* x = in + (size_t)t * stride + off;
    float acc = 0.f;
    for (int i = threadIdx.x; i < len; i += 256) { float v = x[i]; acc += v * v; }
    red[threadIdx.x] = acc;
    __syncthreads();
    for (int s = 128; s > 0; s >>= 1) {
        if (threadIdx.x < s) red[threadIdx.x] += red[threadIdx.x + s];
        __syncthreads();
    }
    float scale = rsqrtf(red[0] / (float)len + 1e-6f);
    for (int i = threadIdx.x; i < len; i += 256)
        out[(size_t)t * len + i] = f2bf(x[i] * scale * w[i]);
}

// ---------------------------------------------------------------------------
// RoPE for k_pe (fp32, kept small): KPE[t][0:64]
// ---------------------------------------------------------------------------
__global__ void k_rope_kpe(const float* __restrict__ QKVL,
                           const int* __restrict__ pos,
                           float* __restrict__ KPE) {
    const int t = blockIdx.x, j = threadIdx.x;  // 32 threads
    const float* src = QKVL + (size_t)t * QKVA_N + (Q_LORA + KV_LORA);
    float p   = (float)pos[t];
    float inv = powf(10000.0f, -(float)j * (1.0f / 32.0f));
    float ang = p * inv, c = cosf(ang), s = sinf(ang);
    float x1 = src[j], x2 = src[32 + j];
    KPE[t * 64 + j]      = x1 * c - x2 * s;
    KPE[t * 64 + 32 + j] = x2 * c + x1 * s;
}

// ---------------------------------------------------------------------------
// Build Q (rope on last 64 dims) -> bf16 [NH][T][192]
// ---------------------------------------------------------------------------
__global__ void k_build_q(const float* __restrict__ Qf,
                          const int* __restrict__ pos,
                          unsigned short* __restrict__ Q16) {
    const int idx = blockIdx.x, t = idx >> 5, h = idx & 31;
    const int d = threadIdx.x;  // 160 threads
    const float* src = Qf + (size_t)t * (NH * DQK) + h * DQK;
    unsigned short* dst = Q16 + ((size_t)h * T_SEQ + t) * DQK;
    if (d < 128) {
        dst[d] = f2bf(src[d]);
    } else {
        int j = d - 128;
        float p   = (float)pos[t];
        float inv = powf(10000.0f, -(float)j * (1.0f / 32.0f));
        float ang = p * inv, c = cosf(ang), s = sinf(ang);
        float x1 = src[128 + j], x2 = src[160 + j];
        dst[128 + j] = f2bf(x1 * c - x2 * s);
        dst[160 + j] = f2bf(x2 * c + x1 * s);
    }
}

// Build K -> bf16 [NH][T][192] (nope from kv_b GEMM, rope broadcast)
__global__ void k_build_k(const float* __restrict__ KVf,
                          const float* __restrict__ KPE,
                          unsigned short* __restrict__ K16) {
    const int idx = blockIdx.x, t = idx >> 5, h = idx & 31;
    const int d = threadIdx.x;  // 192 threads
    unsigned short* dst = K16 + ((size_t)h * T_SEQ + t) * DQK;
    if (d < 128) dst[d] = f2bf(KVf[(size_t)t * (NH * 256) + h * 256 + d]);
    else         dst[d] = f2bf(KPE[t * 64 + (d - 128)]);
}

// Build V transposed -> bf16 [NH][128][T] (so PV B-fragments are contiguous)
__global__ void k_build_v(const float* __restrict__ KVf,
                          unsigned short* __restrict__ Vt) {
    const int idx = blockIdx.x, t = idx >> 5, h = idx & 31;
    const int d = threadIdx.x;  // 128 threads
    Vt[((size_t)h * DV + d) * T_SEQ + t] =
        f2bf(KVf[(size_t)t * (NH * 256) + h * 256 + 128 + d]);
}

// ---------------------------------------------------------------------------
// Causal flash attention, bf16 WMMA, one wave per 16-query tile.
// Q,K: [NH][T][192] bf16; Vt: [NH][128][T] bf16; O: [T][NH*128] bf16.
// K/V working set (~42MB) lives in the 192MB L2.
// ---------------------------------------------------------------------------
__global__ __launch_bounds__(128)
void k_mla_flash(const unsigned short* __restrict__ Q,
                 const unsigned short* __restrict__ Kt,
                 const unsigned short* __restrict__ Vt,
                 unsigned short* __restrict__ O) {
    __shared__ __align__(16) unsigned short Pbuf[4][16 * 32];
    const int lane = threadIdx.x & 31, wave = threadIdx.x >> 5;
    const int lm = lane & 15, lh = lane >> 4;
    const int head  = blockIdx.y;
    const int qbase = blockIdx.x * 64 + wave * 16;

    const unsigned short* Qh = Q  + (size_t)head * T_SEQ * DQK;
    const unsigned short* Kh = Kt + (size_t)head * T_SEQ * DQK;
    const unsigned short* Vh = Vt + (size_t)head * DV * T_SEQ;

    v16bf qf[6];
#pragma unroll
    for (int kk = 0; kk < 6; ++kk)
        qf[kk] = load_frag(&Qh[(size_t)(qbase + lm) * DQK + kk * 32 + lh * 8]);

    f32x8 o[8];
#pragma unroll
    for (int ct = 0; ct < 8; ++ct) o[ct] = fzero8();
    float mrun[8], lrun[8];
#pragma unroll
    for (int e = 0; e < 8; ++e) { mrun[e] = -1e30f; lrun[e] = 0.f; }

    const int kend = qbase + 16;  // causal bound (exclusive)
    for (int ks = 0; ks < kend; ks += 32) {
        f32x8 s[2];
        s[0] = fzero8(); s[1] = fzero8();
#pragma unroll
        for (int c = 0; c < 2; ++c)
#pragma unroll
            for (int kk = 0; kk < 6; ++kk) {
                v16bf kf = load_frag(
                    &Kh[(size_t)(ks + c * 16 + lm) * DQK + kk * 32 + lh * 8]);
                s[c] = __builtin_amdgcn_wmma_f32_16x16x32_bf16(
                    false, qf[kk], false, kf, (short)0, s[c], false, false);
            }
        // Scale + causal mask (C layout: row = qbase+e+8*lh, col = ks+c*16+lm).
#pragma unroll
        for (int c = 0; c < 2; ++c)
#pragma unroll
            for (int e = 0; e < 8; ++e) {
                int row = qbase + e + 8 * lh;
                int col = ks + c * 16 + lm;
                float v = s[c][e] * ATT_SCALE;
                s[c][e] = (col <= row) ? v : -1e30f;
            }
        // Online softmax: row stats reduced across 16-lane halves.
        float fac[8];
#pragma unroll
        for (int e = 0; e < 8; ++e) {
            float rm = fmaxf(s[0][e], s[1][e]);
#pragma unroll
            for (int off = 1; off < 16; off <<= 1)
                rm = fmaxf(rm, __shfl_xor(rm, off, 32));
            float mnew = fmaxf(mrun[e], rm);
            fac[e] = __expf(mrun[e] - mnew);
            float p0 = __expf(s[0][e] - mnew);
            float p1 = __expf(s[1][e] - mnew);
            s[0][e] = p0; s[1][e] = p1;
            float rs = p0 + p1;
#pragma unroll
            for (int off = 1; off < 16; off <<= 1)
                rs += __shfl_xor(rs, off, 32);
            lrun[e] = lrun[e] * fac[e] + rs;
            mrun[e] = mnew;
        }
#pragma unroll
        for (int ct = 0; ct < 8; ++ct)
#pragma unroll
            for (int e = 0; e < 8; ++e) o[ct][e] *= fac[e];

        // C-layout -> A-fragment layout via per-wave LDS staging.
        unsigned short* pb = &Pbuf[wave][0];
#pragma unroll
        for (int c = 0; c < 2; ++c)
#pragma unroll
            for (int e = 0; e < 8; ++e)
                pb[(e + 8 * lh) * 32 + c * 16 + lm] = f2bf(s[c][e]);
        v16bf pf = load_frag(&pb[lm * 32 + lh * 8]);

        // O += P V  (V B-fragments contiguous thanks to transposed layout).
#pragma unroll
        for (int ct = 0; ct < 8; ++ct) {
            v16bf vf = load_frag(&Vh[(size_t)(ct * 16 + lm) * T_SEQ + ks + lh * 8]);
            o[ct] = __builtin_amdgcn_wmma_f32_16x16x32_bf16(
                false, pf, false, vf, (short)0, o[ct], false, false);
        }
    }

#pragma unroll
    for (int e = 0; e < 8; ++e) {
        float inv = 1.0f / lrun[e];
        int row = qbase + e + 8 * lh;
#pragma unroll
        for (int ct = 0; ct < 8; ++ct)
            O[(size_t)row * (NH * DV) + head * DV + ct * 16 + lm] =
                f2bf(o[ct][e] * inv);
    }
}

// ---------------------------------------------------------------------------
// Host launch
// ---------------------------------------------------------------------------
extern "C" void kernel_launch(void* const* d_in, const int* in_sizes, int n_in,
                              void* d_out, int out_size, void* d_ws, size_t ws_size,
                              hipStream_t stream) {
    const int*   positions = (const int*)d_in[0];
    const float* hidden    = (const float*)d_in[1];
    const float* W_qkv_a   = (const float*)d_in[2];
    const float* w_q_a_ln  = (const float*)d_in[3];
    const float* W_q_b     = (const float*)d_in[4];
    const float* w_kv_a_ln = (const float*)d_in[5];
    const float* W_kv_b    = (const float*)d_in[6];
    const float* W_o       = (const float*)d_in[7];
    float* out = (float*)d_out;

    char* w = (char*)d_ws;
    auto alloc = [&](size_t bytes) -> void* {
        void* p = (void*)w;
        w += (bytes + 255) & ~(size_t)255;
        return p;
    };
    unsigned short* X16    = (unsigned short*)alloc((size_t)T_SEQ * HIDDEN * 2);
    unsigned short* WaT    = (unsigned short*)alloc((size_t)QKVA_N * HIDDEN * 2);
    unsigned short* WqbT   = (unsigned short*)alloc((size_t)(NH * DQK) * Q_LORA * 2);
    unsigned short* WkvbT  = (unsigned short*)alloc((size_t)(NH * 256) * KV_LORA * 2);
    unsigned short* WoT    = (unsigned short*)alloc((size_t)HIDDEN * (NH * DV) * 2);
    float*          QKVL   = (float*)alloc((size_t)T_SEQ * QKVA_N * 4);
    unsigned short* QC16   = (unsigned short*)alloc((size_t)T_SEQ * Q_LORA * 2);
    unsigned short* KVC16  = (unsigned short*)alloc((size_t)T_SEQ * KV_LORA * 2);
    float*          KPE    = (float*)alloc((size_t)T_SEQ * QK_ROPE * 4);
    float*          Qf     = (float*)alloc((size_t)T_SEQ * NH * DQK * 4);
    float*          KVf    = (float*)alloc((size_t)T_SEQ * NH * 256 * 4);
    unsigned short* Q16    = (unsigned short*)alloc((size_t)NH * T_SEQ * DQK * 2);
    unsigned short* K16    = (unsigned short*)alloc((size_t)NH * T_SEQ * DQK * 2);
    unsigned short* Vt16   = (unsigned short*)alloc((size_t)NH * DV * T_SEQ * 2);
    unsigned short* A16    = (unsigned short*)alloc((size_t)T_SEQ * NH * DV * 2);

    // 1) activations -> bf16 ; weights -> bf16 TRANSPOSED [N][K]
    k_f32_to_bf16<<<1024, 256, 0, stream>>>(hidden, X16, (long)T_SEQ * HIDDEN);
    k_f32_to_bf16_T<<<dim3(QKVA_N / 32, HIDDEN / 32),  dim3(32, 8), 0, stream>>>(
        W_qkv_a, WaT, HIDDEN, QKVA_N);
    k_f32_to_bf16_T<<<dim3((NH * DQK) / 32, Q_LORA / 32), dim3(32, 8), 0, stream>>>(
        W_q_b, WqbT, Q_LORA, NH * DQK);
    k_f32_to_bf16_T<<<dim3((NH * 256) / 32, KV_LORA / 32), dim3(32, 8), 0, stream>>>(
        W_kv_b, WkvbT, KV_LORA, NH * 256);
    k_f32_to_bf16_T<<<dim3(HIDDEN / 32, (NH * DV) / 32), dim3(32, 8), 0, stream>>>(
        W_o, WoT, NH * DV, HIDDEN);

    // 2) qkv_lora = hidden @ W_qkv_a   [2048 x 2112]
    k_gemm_bf16_tn<<<dim3(T_SEQ / 256, QKVA_N / 64), 256, 0, stream>>>(
        X16, WaT, QKVL, T_SEQ, QKVA_N, HIDDEN);

    // 3) norms + k_pe rope
    k_rmsnorm_bf16<<<T_SEQ, 256, 0, stream>>>(QKVL, QKVA_N, 0,      w_q_a_ln,  QC16,  Q_LORA);
    k_rmsnorm_bf16<<<T_SEQ, 256, 0, stream>>>(QKVL, QKVA_N, Q_LORA, w_kv_a_ln, KVC16, KV_LORA);
    k_rope_kpe<<<T_SEQ, 32, 0, stream>>>(QKVL, positions, KPE);

    // 4) q = q_c_norm @ W_q_b ; kv = kv_c_norm @ W_kv_b
    k_gemm_bf16_tn<<<dim3(T_SEQ / 256, (NH * DQK) / 64), 256, 0, stream>>>(
        QC16, WqbT, Qf, T_SEQ, NH * DQK, Q_LORA);
    k_gemm_bf16_tn<<<dim3(T_SEQ / 256, (NH * 256) / 64), 256, 0, stream>>>(
        KVC16, WkvbT, KVf, T_SEQ, NH * 256, KV_LORA);

    // 5) repack to attention layouts (+ q rope)
    k_build_q<<<T_SEQ * NH, 160, 0, stream>>>(Qf, positions, Q16);
    k_build_k<<<T_SEQ * NH, 192, 0, stream>>>(KVf, KPE, K16);
    k_build_v<<<T_SEQ * NH, 128, 0, stream>>>(KVf, Vt16);

    // 6) causal flash attention
    k_mla_flash<<<dim3(T_SEQ / 64, NH), 128, 0, stream>>>(Q16, K16, Vt16, A16);

    // 7) out = attn @ W_o   [2048 x 4096] fp32
    k_gemm_bf16_tn<<<dim3(T_SEQ / 256, HIDDEN / 64), 256, 0, stream>>>(
        A16, WoT, out, T_SEQ, HIDDEN, NH * DV);
}
